// RelationNetworkModel_28836410426101
// MI455X (gfx1250) — matrix operand order repaired and checked by reading, not compiled, not run on hardware
//
#include <hip/hip_runtime.h>
#include <stdint.h>

// ---------- types ----------
typedef __bf16 v16bf __attribute__((ext_vector_type(16)));
typedef float  f32x8 __attribute__((ext_vector_type(8)));
typedef unsigned int vu32x4 __attribute__((ext_vector_type(4)));
typedef int vi32x8 __attribute__((ext_vector_type(8)));
typedef int vi32x4 __attribute__((ext_vector_type(4)));

union FragBF { vu32x4 q[2]; v16bf v; };

#if __has_builtin(__builtin_amdgcn_tensor_load_to_lds) && __has_builtin(__builtin_amdgcn_s_wait_tensorcnt)
#define USE_TDM 1
#endif

// compile-time introspection (read back via stderr)
#define STR2(x) #x
#define STR(x) STR2(x)
#pragma message("CDNA5 probe: clang major = " STR(__clang_major__))
#ifdef USE_TDM
#pragma message("CDNA5 probe: tensor_load_to_lds builtin AVAILABLE -> TDM weight staging compiled")
#else
#pragma message("CDNA5 probe: tensor_load_to_lds builtin NOT available -> fallback LDS copy path")
#endif

// ---------- fast transcendentals (v_rcp_f32 + v_exp_f32, avoids IEEE div expansion) ----------
__device__ __forceinline__ float fast_rcp(float x) {
#if __has_builtin(__builtin_amdgcn_rcpf)
  return __builtin_amdgcn_rcpf(x);
#else
  return 1.f / x;
#endif
}
__device__ __forceinline__ float fast_sig(float x) { return fast_rcp(1.f + __expf(-x)); }
__device__ __forceinline__ float fast_tanh(float x) {
  return 1.f - 2.f * fast_rcp(__expf(2.f * x) + 1.f);
}

// ---------- WMMA fragment helpers (CDNA5 ISA 7.12.2 layouts, wave32) ----------
// A-matrix 16x32 bf16: lanes 0-15 row M=lane, elements 0..7 = K+0..7, 8..15 = K+16..23;
// lanes 16-31 row M=lane-16, elements 0..7 = K+8..15, 8..15 = K+24..31.
__device__ __forceinline__ v16bf lds_frag_a(const __bf16* tile, int ld) {
  int l = threadIdx.x & 31;
  const __bf16* p = tile + (l & 15) * ld + ((l >> 4) << 3);
  FragBF f;
  f.q[0] = *(const vu32x4*)(p);
  f.q[1] = *(const vu32x4*)(p + 16);
  return f.v;
}
// B-matrix 32x16 bf16 from row-major W[N][K] (B(k,n)=W[n][k]): lane covers N=l%16,
// lanes 0-15 K=kbase+0..7,16..23, lanes 16-31 K=kbase+8..15,24..31 -> two 16B loads.
__device__ __forceinline__ v16bf frag_b_rowmajor(const __bf16* Wtile, int ld) {
  int l = threadIdx.x & 31;
  const __bf16* p = Wtile + (l & 15) * ld + ((l >> 4) << 3);
  FragBF f;
  f.q[0] = *(const vu32x4*)(p);
  f.q[1] = *(const vu32x4*)(p + 16);
  return f.v;
}
__device__ __forceinline__ f32x8 wmma_bf16(v16bf a, v16bf b, f32x8 c) {
  return __builtin_amdgcn_wmma_f32_16x16x32_bf16(false, a, false, b, (short)0, c, false, false);
}
// C/D 16x16 f32: VGPR r -> row r (lanes 0-15) / row 8+r (lanes 16-31), col = lane%16.
__device__ __forceinline__ void store_tile_relu(f32x8 acc, float bv, __bf16* tile, int ld) {
  int l = threadIdx.x & 31;
  int ro = (l >> 4) << 3;
  for (int r = 0; r < 8; ++r) {
    float v = acc[r] + bv;
    v = v > 0.f ? v : 0.f;
    tile[(ro + r) * ld + (l & 15)] = (__bf16)v;
  }
}

#ifdef USE_TDM
// TDM: 2D tile, 256 rows x 256B (data_size=8B, tensor_dim0=tile_dim0=32, row stride 512B),
// LDS pad via D#: pad_interval code 5 (=64 DW = one 256B row), pad_amount code 7 (=8 DW = 32B)
// -> LDS row pitch 288B = our 144-halves wbuf pitch.
__device__ __forceinline__ void tdm_stage_weights(const __bf16* gsrc, __bf16* ldst) {
  unsigned long long ga = (unsigned long long)(size_t)gsrc;
  unsigned int la = (unsigned int)(size_t)ldst;  // generic LDS ptr: low 32 bits = LDS offset
  vu32x4 g0;
  g0[0] = 1u;                                            // count=1
  g0[1] = la;                                            // lds_addr
  g0[2] = (unsigned int)ga;                              // global_addr[31:0]
  g0[3] = (unsigned int)((ga >> 32) & 0x01FFFFFFull) | 0x80000000u;  // addr[56:32] | type=2
  vi32x8 g1;
  g1[0] = (3 << 16) | (1 << 20) | (5 << 22) | (7 << 25); // data_size=8B, pad_en, interval, amount
  g1[1] = (int)(32u << 16);                              // tensor_dim0 (low16 @ bit48)
  g1[2] = (int)(256u << 16);                             // tensor_dim1 (low16 @ bit80)
  g1[3] = (int)(32u << 16);                              // tile_dim0 @ bit112
  g1[4] = 256;                                           // tile_dim1
  g1[5] = 64;                                            // tensor_dim0_stride (=512B/8)
  g1[6] = 0;
  g1[7] = 0;
  vi32x4 z4; z4[0] = z4[1] = z4[2] = z4[3] = 0;
#if __clang_major__ >= 23
  vi32x8 z8; for (int i = 0; i < 8; ++i) z8[i] = 0;
  __builtin_amdgcn_tensor_load_to_lds(g0, g1, z4, z4, z8, 0);
#else
  __builtin_amdgcn_tensor_load_to_lds(g0, g1, z4, z4, 0);
#endif
}
#endif

// ---------- small prep kernels ----------
__global__ void k_zero(float* f_in, float* sum, float* sq) {
  int idx = blockIdx.x * 256 + threadIdx.x;
  if (idx < 8192) f_in[idx] = 0.f;
  if (idx < 96) { sum[idx] = 0.f; sq[idx] = 0.f; }
}

__global__ void k_wcvt(const float* whh, const float* gsW, __bf16* whh_bf, __bf16* gw_bf) {
  int idx = blockIdx.x * 256 + threadIdx.x;
  if (idx < 65536) whh_bf[idx] = (__bf16)whh[idx];
  if (idx < 196608) gw_bf[idx] = (__bf16)gsW[idx];
}

// xW[t,n,g] = emb(q[n,t]) . Wih[g] + bih[g] + bhh[g]   (parallel over all timesteps)
__global__ void k_xw(const int* __restrict__ questions, const float* __restrict__ embW,
                     const float* __restrict__ Wih, const float* __restrict__ bih,
                     const float* __restrict__ bhh, float* __restrict__ xW) {
  int idx = blockIdx.x * 256 + threadIdx.x;
  if (idx >= 46 * 32 * 512) return;
  int g = idx & 511;
  int nt = idx >> 9;
  int n = nt & 31, t = nt >> 5;
  int q = questions[n * 46 + t];
  const float* e = embW + q * 32;
  const float* w = Wih + (size_t)g * 32;
  float a = bih[g] + bhh[g];
  for (int k = 0; k < 32; ++k) a += e[k] * w[k];
  xW[(size_t)(t * 32 + n) * 512 + g] = a;
}

// ---------- persistent single-WGP WMMA LSTM ----------
// Per step: zg(32x512) = xW[t] + h @ Whh^T via v_wmma_f32_16x16x32_bf16 (M=32,N=512,K=128),
// h resident in LDS bf16, Whh streamed bf16 from L2 (two 16B loads per lane).
__global__ void __launch_bounds__(256) k_lstm(const float* __restrict__ xW,
                                              const __bf16* __restrict__ whh_bf,
                                              float* __restrict__ qf) {
  __shared__ __bf16 hls[32 * 136];
  __shared__ float cls[32 * 128];
  __shared__ float zgl[32 * 520];
  const int tid = threadIdx.x;
  const int l = tid & 31, wv = tid >> 5;
  for (int i = tid; i < 32 * 136; i += 256) hls[i] = (__bf16)0.f;
  for (int i = tid; i < 32 * 128; i += 256) cls[i] = 0.f;
  __syncthreads();
  for (int t = 0; t < 46; ++t) {
    const float* xp = xW + (size_t)t * 32 * 512;
    for (int tl = 0; tl < 8; ++tl) {          // 8 output tiles per wave: 2 row x 4 col
      int ct = wv * 4 + (tl & 3);
      int rt = tl >> 2;
      int col = ct * 16 + (l & 15);
      int ro = rt * 16 + ((l >> 4) << 3);
      f32x8 acc;
      for (int r = 0; r < 8; ++r) acc[r] = xp[(size_t)(ro + r) * 512 + col];
      for (int k = 0; k < 4; ++k) {
        v16bf a = lds_frag_a(hls + rt * 16 * 136 + k * 32, 136);
        v16bf b = frag_b_rowmajor(whh_bf + (size_t)(ct * 16) * 128 + k * 32, 128);
        acc = wmma_bf16(a, b, acc);
      }
      for (int r = 0; r < 8; ++r) zgl[(ro + r) * 520 + col] = acc[r];
    }
    __syncthreads();
    for (int idx = tid; idx < 32 * 128; idx += 256) {  // gate nonlinearity (i,f,g,o)
      int n = idx >> 7, d = idx & 127;
      float zi = zgl[n * 520 + d];
      float zf = zgl[n * 520 + 128 + d];
      float zc = zgl[n * 520 + 256 + d];
      float zo = zgl[n * 520 + 384 + d];
      float c = fast_sig(zf) * cls[idx] + fast_sig(zi) * fast_tanh(zc);
      cls[idx] = c;
      hls[n * 136 + d] = (__bf16)(fast_sig(zo) * fast_tanh(c));
    }
    __syncthreads();
  }
  for (int idx = tid; idx < 32 * 128; idx += 256) {
    int n = idx >> 7, d = idx & 127;
    qf[idx] = (float)hls[n * 136 + d];
  }
}

// ---------- conv + batch stats ----------
__global__ void __launch_bounds__(256) k_conv(const float* __restrict__ in,
    const float* __restrict__ Wt, const float* __restrict__ bias,
    float* __restrict__ raw, float* __restrict__ sum, float* __restrict__ sq,
    int Cin, int Hin, int Cout, int Hout) {
  __shared__ float s1[256], s2[256];
  int nb = blockIdx.x;
  int n = nb / Cout, co = nb % Cout;
  int hw = Hout * Hout;
  const float* wb = Wt + (size_t)co * Cin * 9;
  float ps = 0.f, pq = 0.f;
  for (int p = threadIdx.x; p < hw; p += 256) {
    int ho = p / Hout, wo = p % Hout;
    float a = bias[co];
    for (int ci = 0; ci < Cin; ++ci) {
      const float* ip = in + ((size_t)(n * Cin + ci)) * Hin * Hin;
      const float* wp = wb + ci * 9;
      for (int kh = 0; kh < 3; ++kh) {
        int hi = ho * 2 - 1 + kh;
        if (hi < 0 || hi >= Hin) continue;
        for (int kw = 0; kw < 3; ++kw) {
          int wi = wo * 2 - 1 + kw;
          if (wi < 0 || wi >= Hin) continue;
          a += ip[hi * Hin + wi] * wp[kh * 3 + kw];
        }
      }
    }
    raw[(size_t)nb * hw + p] = a;
    ps += a; pq += a * a;
  }
  s1[threadIdx.x] = ps; s2[threadIdx.x] = pq;
  __syncthreads();
  for (int s = 128; s > 0; s >>= 1) {
    if (threadIdx.x < s) { s1[threadIdx.x] += s1[threadIdx.x + s]; s2[threadIdx.x] += s2[threadIdx.x + s]; }
    __syncthreads();
  }
  if (threadIdx.x == 0) { atomicAdd(&sum[co], s1[0]); atomicAdd(&sq[co], s2[0]); }
}

__global__ void k_bnstat(const float* sum, const float* sq, const float* gamma,
                         const float* beta, float* scale, float* shift, int cnt) {
  int c = threadIdx.x;
  if (c < 24) {
    float mu = sum[c] / (float)cnt;
    float var = sq[c] / (float)cnt - mu * mu;
    float sc = gamma[c] * rsqrtf(var + 1e-5f);
    scale[c] = sc;
    shift[c] = beta[c] - mu * sc;
  }
}

__global__ void k_bnrelu(const float* __restrict__ raw, const float* __restrict__ scale,
                         const float* __restrict__ shift, float* __restrict__ outp,
                         int hw, int total) {
  int idx = blockIdx.x * 256 + threadIdx.x;
  if (idx >= total) return;
  int c = (idx / hw) % 24;
  float v = raw[idx] * scale[c] + shift[c];
  outp[idx] = v > 0.f ? v : 0.f;
}

// ---------- object / pair-part precompute (g0 column split) ----------
__global__ void k_objprep(const float* __restrict__ act3, const float* __restrict__ g0W,
                          const float* __restrict__ g0b, const float* __restrict__ qf,
                          float* __restrict__ Apart, float* __restrict__ Bpart,
                          float* __restrict__ Qb) {
  int idx = blockIdx.x * 256 + threadIdx.x;
  if (idx >= 32 * 64 * 256) return;
  int o = idx & 255;
  int ni = idx >> 8;
  int i = ni & 63, n = ni >> 6;
  int h = i >> 3, w = i & 7;
  float obj[26];
  for (int c = 0; c < 24; ++c) obj[c] = act3[(((size_t)(n * 24 + c)) * 8 + h) * 8 + w];
  obj[24] = (float)w * (1.f / 7.f);
  obj[25] = (float)h * (1.f / 7.f);
  const float* wr = g0W + (size_t)o * 180;
  float a = 0.f, b = 0.f;
  for (int c = 0; c < 26; ++c) { a += obj[c] * wr[c]; b += obj[c] * wr[26 + c]; }
  Apart[idx] = a;
  Bpart[idx] = b;
  if (i == 0) {
    float qv = g0b[o];
    const float* qrow = qf + n * 128;
    for (int k = 0; k < 128; ++k) qv += qrow[k] * wr[52 + k];
    Qb[n * 256 + o] = qv;
  }
}

// ---------- fused g-MLP: h0 build + 3x WMMA(256x256) + pair-sum reduction ----------
// Block = 64 pairs x 256 features; 8 waves = 4 row-tiles x 2 col-halves, acc[8] per wave.
// Weights staged per k-half into LDS via TDM (padded 288B pitch), activations ping-pong in LDS.
__global__ void __launch_bounds__(256) k_gfused(const float* __restrict__ Apart,
    const float* __restrict__ Bpart, const float* __restrict__ Qb,
    const __bf16* __restrict__ gwbf, const float* __restrict__ gs_b,
    float* __restrict__ f_in) {
  __shared__ __bf16 actA[64 * 272];
  __shared__ __bf16 actB[64 * 272];
  __shared__ __bf16 wlds[256 * 144];
  const int tid = threadIdx.x;
  const int l = tid & 31, wv = tid >> 5;
  const int n = blockIdx.x >> 6;
  const int tile = blockIdx.x & 63;
  // stage 0: h0 = relu(Apart_i + Bpart_j + Qb)
  const float* qb = Qb + n * 256;
  for (int idx = tid; idx < 64 * 256; idx += 256) {
    int r = idx >> 8, c = idx & 255;
    int p = tile * 64 + r;
    int i = p >> 6, j = p & 63;
    float v = Apart[(size_t)(n * 64 + i) * 256 + c] +
              Bpart[(size_t)(n * 64 + j) * 256 + c] + qb[c];
    actA[r * 272 + c] = (__bf16)(v > 0.f ? v : 0.f);
  }
  const int rt = wv & 3;
  const int ctbase = (wv >> 2) * 8;
  __bf16* aIn = actA;
  __bf16* aOut = actB;
  for (int layer = 0; layer < 3; ++layer) {
    const __bf16* W = gwbf + (size_t)layer * 256 * 256;
    const float* bias = gs_b + layer * 256;
    f32x8 acc[8];
    for (int c = 0; c < 8; ++c)
      for (int r = 0; r < 8; ++r) acc[c][r] = 0.f;
    for (int kh = 0; kh < 2; ++kh) {
      __syncthreads();                       // all waves done with previous wlds contents
#ifdef USE_TDM
      if (tid == 0) tdm_stage_weights(W + kh * 128, wlds);
      __builtin_amdgcn_s_wait_tensorcnt(0);
#else
      for (int ch = tid; ch < 4096; ch += 256) {
        int rr = ch >> 4, cc = ch & 15;
        *(vu32x4*)&wlds[rr * 144 + cc * 8] =
            *(const vu32x4*)&W[(size_t)rr * 256 + kh * 128 + cc * 8];
      }
#endif
      __syncthreads();
      for (int k = 0; k < 4; ++k) {
        v16bf a = lds_frag_a(aIn + rt * 16 * 272 + kh * 128 + k * 32, 272);
        for (int c = 0; c < 8; ++c) {
          v16bf b = frag_b_rowmajor(wlds + (ctbase + c) * 16 * 144 + k * 32, 144);
          acc[c] = wmma_bf16(a, b, acc[c]);
        }
      }
    }
    if (layer < 2) {
      for (int c = 0; c < 8; ++c)
        store_tile_relu(acc[c], bias[(ctbase + c) * 16 + (l & 15)],
                        aOut + rt * 16 * 272 + (ctbase + c) * 16, 272);
      __bf16* tmp = aIn; aIn = aOut; aOut = tmp;
    } else {
      // relu + sum over the 16 rows of each tile, fold wave halves, atomic into f_in[n]
      for (int c = 0; c < 8; ++c) {
        float bv = bias[(ctbase + c) * 16 + (l & 15)];
        float partial = 0.f;
        for (int r = 0; r < 8; ++r) {
          float v = acc[c][r] + bv;
          partial += v > 0.f ? v : 0.f;
        }
        partial += __shfl_xor(partial, 16, 32);
        if (l < 16) atomicAdd(&f_in[n * 256 + (ctbase + c) * 16 + l], partial);
      }
    }
  }
}

// ---------- f-MLP (tiny) ----------
__global__ void k_dense(const float* __restrict__ x, const float* __restrict__ W,
                        const float* __restrict__ b, float* __restrict__ y,
                        int rows, int K, int O, int relu) {
  int idx = blockIdx.x * 256 + threadIdx.x;
  if (idx >= rows * O) return;
  int o = idx % O, r = idx / O;
  const float* xr = x + (size_t)r * K;
  const float* wr = W + (size_t)o * K;
  float a = b[o];
  for (int k = 0; k < K; ++k) a += xr[k] * wr[k];
  y[idx] = relu ? (a > 0.f ? a : 0.f) : a;
}

// ---------- workspace layout (bytes) ----------
static constexpr size_t OFF_WHH = 0;                       // 512*128*2
static constexpr size_t OFF_GW  = OFF_WHH + 131072;        // 3*256*256*2
static constexpr size_t OFF_XW  = OFF_GW + 393216;         // 46*32*512*4
static constexpr size_t OFF_QF  = OFF_XW + 3014656;        // 32*128*4
static constexpr size_t OFF_RAW = OFF_QF + 16384;          // 32*24*64*64*4
static constexpr size_t OFF_A0  = OFF_RAW + 12582912;
static constexpr size_t OFF_A1  = OFF_A0 + 12582912;
static constexpr size_t OFF_A2  = OFF_A1 + 3145728;
static constexpr size_t OFF_A3  = OFF_A2 + 786432;
static constexpr size_t OFF_SUM = OFF_A3 + 196608;         // 4*24*4
static constexpr size_t OFF_SQ  = OFF_SUM + 384;
static constexpr size_t OFF_SC  = OFF_SQ + 384;
static constexpr size_t OFF_SH  = OFF_SC + 384;
static constexpr size_t OFF_AP  = OFF_SH + 384;            // 2048*256*4
static constexpr size_t OFF_BP  = OFF_AP + 2097152;
static constexpr size_t OFF_QB  = OFF_BP + 2097152;        // 32*256*4
static constexpr size_t OFF_FIN = OFF_QB + 32768;
static constexpr size_t OFF_FH0 = OFF_FIN + 32768;
static constexpr size_t OFF_FH1 = OFF_FH0 + 32768;

extern "C" void kernel_launch(void* const* d_in, const int* in_sizes, int n_in,
                              void* d_out, int out_size, void* d_ws, size_t ws_size,
                              hipStream_t stream) {
  const int*   questions = (const int*)d_in[0];
  const float* images    = (const float*)d_in[1];
  const float* embed_W   = (const float*)d_in[2];
  const float* lstm_Wih  = (const float*)d_in[3];
  const float* lstm_Whh  = (const float*)d_in[4];
  const float* lstm_bih  = (const float*)d_in[5];
  const float* lstm_bhh  = (const float*)d_in[6];
  const float* conv0_W   = (const float*)d_in[7];
  const float* conv0_b   = (const float*)d_in[8];
  const float* bn0_g     = (const float*)d_in[9];
  const float* bn0_b     = (const float*)d_in[10];
  const float* convs_W   = (const float*)d_in[11];
  const float* convs_b   = (const float*)d_in[12];
  const float* bns_g     = (const float*)d_in[13];
  const float* bns_b     = (const float*)d_in[14];
  const float* g0_W      = (const float*)d_in[15];
  const float* g0_b      = (const float*)d_in[16];
  const float* gs_W      = (const float*)d_in[17];
  const float* gs_bp     = (const float*)d_in[18];
  const float* f0_W      = (const float*)d_in[19];
  const float* f0_b      = (const float*)d_in[20];
  const float* f1_W      = (const float*)d_in[21];
  const float* f1_b      = (const float*)d_in[22];
  const float* f2_W      = (const float*)d_in[23];
  const float* f2_b      = (const float*)d_in[24];
  float* out = (float*)d_out;
  char* ws = (char*)d_ws;

  __bf16* whh_bf = (__bf16*)(ws + OFF_WHH);
  __bf16* gw_bf  = (__bf16*)(ws + OFF_GW);
  float* xW   = (float*)(ws + OFF_XW);
  float* qf   = (float*)(ws + OFF_QF);
  float* raw  = (float*)(ws + OFF_RAW);
  float* a0   = (float*)(ws + OFF_A0);
  float* a1   = (float*)(ws + OFF_A1);
  float* a2   = (float*)(ws + OFF_A2);
  float* a3   = (float*)(ws + OFF_A3);
  float* sum  = (float*)(ws + OFF_SUM);
  float* sq   = (float*)(ws + OFF_SQ);
  float* sc   = (float*)(ws + OFF_SC);
  float* sh   = (float*)(ws + OFF_SH);
  float* Ap   = (float*)(ws + OFF_AP);
  float* Bp   = (float*)(ws + OFF_BP);
  float* Qb   = (float*)(ws + OFF_QB);
  float* fin  = (float*)(ws + OFF_FIN);
  float* fh0  = (float*)(ws + OFF_FH0);
  float* fh1  = (float*)(ws + OFF_FH1);

  k_zero<<<32, 256, 0, stream>>>(fin, sum, sq);
  k_wcvt<<<768, 256, 0, stream>>>(lstm_Whh, gs_W, whh_bf, gw_bf);
  k_xw<<<2944, 256, 0, stream>>>(questions, embed_W, lstm_Wih, lstm_bih, lstm_bhh, xW);
  k_lstm<<<1, 256, 0, stream>>>(xW, whh_bf, qf);

  // conv stack: (Cin,Hin)->(24,Hout), raw buffer reused per layer
  k_conv<<<768, 256, 0, stream>>>(images, conv0_W, conv0_b, raw, sum + 0, sq + 0, 3, 128, 24, 64);
  k_bnstat<<<1, 32, 0, stream>>>(sum + 0, sq + 0, bn0_g, bn0_b, sc + 0, sh + 0, 32 * 64 * 64);
  k_bnrelu<<<(32 * 24 * 64 * 64 + 255) / 256, 256, 0, stream>>>(raw, sc + 0, sh + 0, a0, 64 * 64, 32 * 24 * 64 * 64);

  const float* cin[3] = { a0, a1, a2 };
  float* cout_[3] = { a1, a2, a3 };
  int hin[3] = { 64, 32, 16 };
  for (int i = 0; i < 3; ++i) {
    int Hout = hin[i] / 2;
    int hw = Hout * Hout;
    int total = 32 * 24 * hw;
    k_conv<<<768, 256, 0, stream>>>(cin[i], convs_W + (size_t)i * 24 * 24 * 9, convs_b + i * 24,
                                    raw, sum + (i + 1) * 24, sq + (i + 1) * 24, 24, hin[i], 24, Hout);
    k_bnstat<<<1, 32, 0, stream>>>(sum + (i + 1) * 24, sq + (i + 1) * 24, bns_g + i * 24,
                                   bns_b + i * 24, sc + (i + 1) * 24, sh + (i + 1) * 24, 32 * hw);
    k_bnrelu<<<(total + 255) / 256, 256, 0, stream>>>(raw, sc + (i + 1) * 24, sh + (i + 1) * 24,
                                                      cout_[i], hw, total);
  }

  k_objprep<<<2048, 256, 0, stream>>>(a3, g0_W, g0_b, qf, Ap, Bp, Qb);
  k_gfused<<<2048, 256, 0, stream>>>(Ap, Bp, Qb, gw_bf, gs_bp, fin);

  k_dense<<<32, 256, 0, stream>>>(fin, f0_W, f0_b, fh0, 32, 256, 256, 1);
  k_dense<<<32, 256, 0, stream>>>(fh0, f1_W, f1_b, fh1, 32, 256, 256, 1);
  k_dense<<<4, 256, 0, stream>>>(fh1, f2_W, f2_b, out, 32, 256, 32, 0);
  (void)in_sizes; (void)n_in; (void)out_size; (void)ws_size;
}
